// NodeEdgeBlock_8203387535990
// MI455X (gfx1250) — matrix operand order repaired and checked by reading, not compiled, not run on hardware
//
#include <hip/hip_runtime.h>

// ---------------------------------------------------------------------------
// CDNA5 (gfx1250) implementation. wave32, WMMA bf16 16x16x32 with f32 accum.
// Fragment-major LDS staging: each lane's 16 bf16 operand values are one
// contiguous v16bf -> wide ds loads instead of 32x ds_load_u16 per fragment.
// K-permutation trick: WMMA sums over K, so A and B may use any shared
// permutation of K slots; A and B here both stage contiguous k order
// (lane m+16h holds k = 16h..16h+15), not AMD's table order; bperm()'s result is unused.
// ---------------------------------------------------------------------------

typedef __attribute__((ext_vector_type(16))) __bf16 v16bf;
typedef __attribute__((ext_vector_type(8)))  float  v8f;

__device__ inline v8f v8zero() {
  v8f z = {0.f,0.f,0.f,0.f,0.f,0.f,0.f,0.f};
  return z;
}

__device__ inline v8f wmma_bf16(v16bf a, v16bf b, v8f c) {
  return __builtin_amdgcn_wmma_f32_16x16x32_bf16(false, a, false, b, (short)0, c,
                                                 false, false);
}

// Involution swapping k 8..15 <-> 16..23; k_wvh computes it but stages B by kr.
__device__ inline int bperm(int k) {
  int g = (k >> 3) & 3;
  return (g == 1 || g == 2) ? (k ^ 24) : k;
}

// ---------------------------------------------------------------------------
// Generic bf16-WMMA GEMM: C[M,N] = A[M,K] * W[N,K]^T (+bias)(relu?)(*rowmask)
// grid = (M/64, N/64), block = 256 (8 waves, 2 16x16 tiles each).
// ---------------------------------------------------------------------------
__global__ __launch_bounds__(256)
void k_gemm(const float* __restrict__ A, const float* __restrict__ W,
            const float* __restrict__ bias, const float* __restrict__ rowmask,
            float* __restrict__ C, int M, int N, int K, int relu)
{
  __shared__ v16bf sAf[4][32];
  __shared__ v16bf sWf[4][32];
  const int tid = threadIdx.x, wave = tid >> 5, lane = tid & 31;
  const int hf = lane >> 4, ln = lane & 15;
  const int m0 = blockIdx.x * 64, n0 = blockIdx.y * 64;
  const int mt0 = wave & 3, nt0 = wave >> 2, nt1 = (wave >> 2) + 2;
  const int rr = tid >> 2, c0 = (tid & 3) * 8;
  __bf16* dA = (__bf16*)&sAf[rr >> 4][(rr & 15) + ((c0 >> 4) << 4)] + (c0 & 15);
  __bf16* dW = (__bf16*)&sWf[rr >> 4][(rr & 15) + ((c0 >> 4) << 4)] + (c0 & 15);
  v8f acc0 = v8zero(), acc1 = v8zero();
  for (int kk = 0; kk < K; kk += 32) {
    __syncthreads();
    const float* as = &A[(long)(m0+rr)*K + kk + c0];
    const float* ws = &W[(long)(n0+rr)*K + kk + c0];
#pragma unroll
    for (int u = 0; u < 8; ++u) dA[u] = (__bf16)as[u];
#pragma unroll
    for (int u = 0; u < 8; ++u) dW[u] = (__bf16)ws[u];
    __syncthreads();
    acc0 = wmma_bf16(sAf[mt0][lane], sWf[nt0][lane], acc0);
    acc1 = wmma_bf16(sAf[mt0][lane], sWf[nt1][lane], acc1);
  }
  v8f accs[2] = {acc0, acc1};
  const int nts[2] = {nt0, nt1};
#pragma unroll
  for (int t = 0; t < 2; ++t) {
#pragma unroll
    for (int v = 0; v < 8; ++v) {
      int m = m0 + mt0*16 + hf*8 + v;
      int nn = n0 + nts[t]*16 + ln;
      float x = accs[t][v];
      if (bias)    x += bias[nn];
      if (relu)    x = fmaxf(x, 0.f);
      if (rowmask) x *= rowmask[m];
      C[(long)m*N + nn] = x;
    }
  }
}

// ---------------------------------------------------------------------------
// Small per-batch y-derived vectors + node-mask counts.
// ---------------------------------------------------------------------------
__global__ __launch_bounds__(256)
void k_prep(const float* __restrict__ y, const float* __restrict__ mask,
            const float* WyeM, const float* byeM,
            const float* WyeA, const float* byeA,
            const float* Wyxm, const float* byxm,
            const float* Wyxa, const float* byxa,
            const float* Wyy,  const float* byy,
            float* yeM, float* yeA, float* yxm, float* yxa, float* yy,
            float* divA)
{
  int tid = threadIdx.x;
  {
    int b = tid >> 6, c = tid & 63;
    float aM = byeM[c], aA = byeA[c], aY = byy[c];
    for (int k = 0; k < 64; ++k) {
      float yv = y[b*64+k];
      aM += yv * WyeM[c*64+k];
      aA += yv * WyeA[c*64+k];
      aY += yv * Wyy [c*64+k];
    }
    yeM[b*64+c] = aM; yeA[b*64+c] = aA; yy[b*64+c] = aY;
  }
  for (int r = 0; r < 4; ++r) {
    int idx = r*256 + tid; int b = idx >> 8, c = idx & 255;
    float am = byxm[c], aa = byxa[c];
    for (int k = 0; k < 64; ++k) {
      float yv = y[b*64+k];
      am += yv * Wyxm[c*64+k];
      aa += yv * Wyxa[c*64+k];
    }
    yxm[b*256+c] = am; yxa[b*256+c] = aa;
  }
  if (tid < 4) {
    float s = 0.f;
    for (int i = 0; i < 256; ++i) s += mask[tid*256+i];
    divA[tid] = s;
  }
}

// Qh[b][h][i][d] = Q[b,i,d] * Wa[h,d] / sqrt(64)
__global__ __launch_bounds__(256)
void k_qh(const float* __restrict__ Q, const float* __restrict__ Wa,
          float* __restrict__ Qh)
{
  int bi = blockIdx.x; int b = bi >> 8, i = bi & 255;
#pragma unroll
  for (int r = 0; r < 8; ++r) {
    int idx = r*256 + threadIdx.x; int h = idx >> 8, d = idx & 255;
    Qh[(((long)(b*8+h))*256 + i)*256 + d] = Q[(long)bi*256+d]*Wa[h*256+d]*0.125f;
  }
}

// ---------------------------------------------------------------------------
// Raw attention logits: a0[b,i,j,h] = sum_d Qh[b,h,i,d] * K[b,j,d]  (WMMA)
// grid (4 itile, 8 h, 4 b)
// ---------------------------------------------------------------------------
__global__ __launch_bounds__(256)
void k_qk(const float* __restrict__ Qh, const float* __restrict__ Km,
          float* __restrict__ logits)
{
  __shared__ v16bf sAf[4][32];
  __shared__ v16bf sBf[16][32];
  const int tid = threadIdx.x, wave = tid >> 5, lane = tid & 31;
  const int hf = lane >> 4, ln = lane & 15;
  const int mt4 = blockIdx.x, h = blockIdx.y, b = blockIdx.z;
  const float* Ab = Qh + ((long)(b*8+h))*65536 + (long)mt4*64*256;
  const int rr = tid >> 2, c0 = (tid & 3) * 8;
  __bf16* dA = (__bf16*)&sAf[rr >> 4][(rr & 15) + ((c0 >> 4) << 4)] + (c0 & 15);
  v8f acc[8];
#pragma unroll
  for (int s = 0; s < 8; ++s) acc[s] = v8zero();
  for (int kk = 0; kk < 256; kk += 32) {
    __syncthreads();
    const float* as = &Ab[(long)rr*256 + kk + c0];
#pragma unroll
    for (int u = 0; u < 8; ++u) dA[u] = (__bf16)as[u];
#pragma unroll
    for (int r4 = 0; r4 < 4; ++r4) {
      int nr = r4*64 + rr;
      const float* bs = &Km[((long)(b*256)+nr)*256 + kk + c0];
      __bf16* dB = (__bf16*)&sBf[nr >> 4][(nr & 15) + ((c0 >> 4) << 4)] + (c0 & 15);
#pragma unroll
      for (int u = 0; u < 8; ++u) dB[u] = (__bf16)bs[u];
    }
    __syncthreads();
#pragma unroll
    for (int s = 0; s < 8; ++s) {
      int id = wave*8 + s; int mt = id >> 4, nt = id & 15;
      acc[s] = wmma_bf16(sAf[mt][lane], sBf[nt][lane], acc[s]);
    }
  }
#pragma unroll
  for (int s = 0; s < 8; ++s) {
    int id = wave*8 + s; int mt = id >> 4, nt = id & 15;
#pragma unroll
    for (int v = 0; v < 8; ++v) {
      int i = mt4*64 + mt*16 + hf*8 + v;
      int j = nt*16 + ln;
      logits[(((long)(b*256)+i)*256 + j)*8 + h] = acc[s][v];
    }
  }
}

// ---------------------------------------------------------------------------
// wV[b,i,h*256+d] = sum_j alpha[b,i,j,h] * V[b,j,d]   (WMMA)
// ---------------------------------------------------------------------------
__global__ __launch_bounds__(256)
void k_wvh(const float* __restrict__ logits, const float* __restrict__ V,
           float* __restrict__ wV)
{
  __shared__ v16bf sAf[4][32];
  __shared__ v16bf sBf[16][32];
  const int tid = threadIdx.x, wave = tid >> 5, lane = tid & 31;
  const int hf = lane >> 4, ln = lane & 15;
  const int mt4 = blockIdx.x, h = blockIdx.y, b = blockIdx.z;
  const int m0 = mt4 * 64;
  const int rr = tid >> 2, c0 = (tid & 3) * 8;
  __bf16* dA = (__bf16*)&sAf[rr >> 4][(rr & 15) + ((c0 >> 4) << 4)] + (c0 & 15);
  const int kr = tid >> 3, nb = (tid & 7) * 32;
  const int lb = bperm(kr);
  v8f acc[8];
#pragma unroll
  for (int s = 0; s < 8; ++s) acc[s] = v8zero();
  for (int kk = 0; kk < 256; kk += 32) {
    __syncthreads();
#pragma unroll
    for (int u = 0; u < 8; ++u)
      dA[u] = (__bf16)logits[(((long)(b*256)+m0+rr)*256 + kk + c0 + u)*8 + h];
    const float* vs = &V[((long)(b*256)+kk+kr)*256 + nb];
#pragma unroll
    for (int u = 0; u < 32; ++u) {
      int n = nb + u;
      ((__bf16*)&sBf[n >> 4][(n & 15) + ((kr >> 4) << 4)])[kr & 15] = (__bf16)vs[u];
    }
    __syncthreads();
#pragma unroll
    for (int s = 0; s < 8; ++s) {
      int id = wave*8 + s; int mt = id >> 4, nt = id & 15;
      acc[s] = wmma_bf16(sAf[mt][lane], sBf[nt][lane], acc[s]);
    }
  }
#pragma unroll
  for (int s = 0; s < 8; ++s) {
    int id = wave*8 + s; int mt = id >> 4, nt = id & 15;
#pragma unroll
    for (int v = 0; v < 8; ++v) {
      int i = m0 + mt*16 + hf*8 + v;
      int d = nt*16 + ln;
      wV[((long)(b*256)+i)*2048 + h*256 + d] = acc[s][v];
    }
  }
}

// ---------------------------------------------------------------------------
// Edge megakernel helpers.
// ---------------------------------------------------------------------------
__device__ inline void stat_accum(float s, float q, float mn, float mx, int base,
                                  float* sRed, float (*sStat)[12], int tid)
{
  sRed[tid] = s;  __syncthreads();
  if (tid < 64) sStat[tid][base+0] += sRed[tid]+sRed[tid+64]+sRed[tid+128]+sRed[tid+192];
  __syncthreads();
  sRed[tid] = q;  __syncthreads();
  if (tid < 64) sStat[tid][base+1] += sRed[tid]+sRed[tid+64]+sRed[tid+128]+sRed[tid+192];
  __syncthreads();
  sRed[tid] = mn; __syncthreads();
  if (tid < 64) sStat[tid][base+2] =
      fminf(sStat[tid][base+2],
            fminf(fminf(sRed[tid],sRed[tid+64]), fminf(sRed[tid+128],sRed[tid+192])));
  __syncthreads();
  sRed[tid] = mx; __syncthreads();
  if (tid < 64) sStat[tid][base+3] =
      fmaxf(sStat[tid][base+3],
            fmaxf(fmaxf(sRed[tid],sRed[tid+64]), fmaxf(sRed[tid+128],sRed[tid+192])));
  __syncthreads();
}

template <typename FS, typename FF>
__device__ inline void edge_gemm(const float* __restrict__ Wg,
                                 v16bf (*sAf)[32], v16bf (*sWf)[32],
                                 int tid, int lane,
                                 int mtA, int ntA, int mtB, int ntB,
                                 FS stage, FF fini)
{
  v8f a0 = v8zero(), a1 = v8zero();
  const int rr = tid >> 2, c0 = (tid & 3) * 8;
  __bf16* dA = (__bf16*)&sAf[rr >> 4][(rr & 15) + ((c0 >> 4) << 4)] + (c0 & 15);
  __bf16* dW = (__bf16*)&sWf[rr >> 4][(rr & 15) + ((c0 >> 4) << 4)] + (c0 & 15);
  for (int kk = 0; kk < 64; kk += 32) {
    __syncthreads();
#pragma unroll
    for (int u = 0; u < 8; ++u) dA[u] = stage(rr, kk + c0 + u);
    const float* ws = &Wg[rr*64 + kk + c0];
#pragma unroll
    for (int u = 0; u < 8; ++u) dW[u] = (__bf16)ws[u];
    __syncthreads();
    a0 = wmma_bf16(sAf[mtA][lane], sWf[ntA][lane], a0);
    a1 = wmma_bf16(sAf[mtB][lane], sWf[ntB][lane], a1);
  }
  __syncthreads();
  fini(a0, mtA, ntA);
  fini(a1, mtB, ntB);
}

// ---------------------------------------------------------------------------
// Edge megakernel: one block per (b,i) row; chunks of 64 j.
// 5 WMMA GEMMs per chunk; Eout, logit modulation, stats, vel messages.
// ---------------------------------------------------------------------------
__global__ __launch_bounds__(256)
void k_edge(const float* __restrict__ E, const float* __restrict__ pos,
            const float* __restrict__ mask,
            const float* __restrict__ xe1, const float* __restrict__ xe2v,
            const float* __restrict__ W_inE, const float* __restrict__ b_inE,
            const float* __restrict__ W_dd,  const float* __restrict__ b_dd,
            const float* __restrict__ W_da,  const float* __restrict__ b_da,
            const float* __restrict__ W_dm,  const float* __restrict__ b_dm,
            const float* __restrict__ W_eo,  const float* __restrict__ b_eo,
            const float* __restrict__ W_ea,  const float* __restrict__ b_ea,
            const float* __restrict__ W_pa,  const float* __restrict__ b_pa,
            const float* __restrict__ W_p1,  const float* __restrict__ W_p2,
            const float* __restrict__ yeA,   const float* __restrict__ yeM,
            float* __restrict__ Eout, float* __restrict__ logits,
            float* __restrict__ rowStats, float* __restrict__ velRaw)
{
  const int bi = blockIdx.x;
  const int b = bi >> 8;
  const int tid = threadIdx.x, wave = tid >> 5, lane = tid & 31;
  const int hf = lane >> 4, ln = lane & 15;

  __shared__ float  sY[64][64];
  __shared__ float  sD[64][64];
  __shared__ v16bf  sAf[4][32];
  __shared__ v16bf  sWf[4][32];
  __shared__ float  sWd[64][8];
  __shared__ float  sWatt[16][64];
  __shared__ float  sAttB[16];
  __shared__ float  sVec[8][64];
  __shared__ float  sMaskj[64];
  __shared__ float  sPosj[64][3];
  __shared__ float  sStat[64][12];
  __shared__ float  sMsgP[64][4];
  __shared__ float  sMsg[64];
  __shared__ float  sRed[256];

  const float mi = mask[bi];
  const float pi0 = pos[bi*3+0]*mi, pi1 = pos[bi*3+1]*mi, pi2 = pos[bi*3+2]*mi;

  if (tid < 64) {
    int c = tid;
    for (int t = 0; t < 6; ++t) sWd[c][t] = W_dd[c*6+t];
    sWd[c][6] = b_dd[c];
    sVec[0][c] = xe2v[(long)bi*64 + c];
    sVec[1][c] = yeA[b*64+c];
    sVec[2][c] = yeM[b*64+c];
    sVec[3][c] = W_p2[c];
    sVec[4][c] = b_inE[c];
    sVec[5][c] = b_da[c];
    sVec[6][c] = b_dm[c];
    sVec[7][c] = b_eo[c];
    for (int s = 0; s < 12; ++s)
      sStat[c][s] = ((s&3)==2) ? 1e30f : (((s&3)==3) ? -1e30f : 0.f);
  }
  for (int r = 0; r < 4; ++r) {
    int idx = r*256 + tid; int hh = idx >> 6, cc = idx & 63;
    sWatt[hh][cc] = (hh < 8) ? W_ea[hh*64+cc] : W_pa[(hh-8)*64+cc];
  }
  if (tid < 16) sAttB[tid] = (tid < 8) ? b_ea[tid] : b_pa[tid-8];

  const int mtA = wave & 3, ntA = wave >> 2;
  const int mtB = wave & 3, ntB = (wave >> 2) + 2;
  float velAcc = 0.f;
  const float pia = (tid == 0) ? pi0 : ((tid == 1) ? pi1 : pi2);

  for (int jc = 0; jc < 256; jc += 64) {
    __syncthreads();
    if (tid < 64) {
      float mj = mask[b*256 + jc + tid];
      sMaskj[tid] = mj;
      sPosj[tid][0] = pos[(b*256+jc+tid)*3+0]*mj;
      sPosj[tid][1] = pos[(b*256+jc+tid)*3+1]*mj;
      sPosj[tid][2] = pos[(b*256+jc+tid)*3+2]*mj;
    }
    __syncthreads();
    // dist1 = relu(lin6->64(pair)) * emask
#pragma unroll 4
    for (int r = 0; r < 16; ++r) {
      int idx = r*256 + tid; int j = idx >> 6, c = idx & 63;
      float acc = sWd[c][6]
        + sWd[c][0]*pi0 + sWd[c][1]*pi1 + sWd[c][2]*pi2
        + sWd[c][3]*sPosj[j][0] + sWd[c][4]*sPosj[j][1] + sWd[c][5]*sPosj[j][2];
      sD[j][c] = fmaxf(acc, 0.f) * sMaskj[j] * mi;
    }

    const long eBase  = (((long)bi)*256 + jc)*64;
    // GEMM1: Y0 = E * W_inE^T + b
    {
      auto st = [&](int rr, int col) -> __bf16 {
        return (__bf16)E[eBase + (long)rr*64 + col];
      };
      auto fn = [&](v8f acc, int mt, int nt) {
#pragma unroll
        for (int v = 0; v < 8; ++v) {
          int m = mt*16 + hf*8 + v, n = nt*16 + ln;
          sY[m][n] = acc[v] + sVec[4][n];
        }
      };
      edge_gemm(W_inE, sAf, sWf, tid, lane, mtA,ntA,mtB,ntB, st, fn);
    }
    __syncthreads();
    // Y *= xe1[j] * xe2[i] * emask
#pragma unroll 4
    for (int r = 0; r < 16; ++r) {
      int idx = r*256 + tid; int j = idx >> 6, c = idx & 63;
      float em = sMaskj[j] * mi;
      sY[j][c] *= xe1[((long)(b*256)+jc+j)*64 + c] * sVec[0][c] * em;
    }
    // GEMM2: Y *= (1 + dist1*W_dm^T + b_dm)
    {
      auto st = [&](int rr, int col) -> __bf16 { return (__bf16)sD[rr][col]; };
      auto fn = [&](v8f acc, int mt, int nt) {
#pragma unroll
        for (int v = 0; v < 8; ++v) {
          int m = mt*16 + hf*8 + v, n = nt*16 + ln;
          sY[m][n] *= (1.f + acc[v] + sVec[6][n]);
        }
      };
      edge_gemm(W_dm, sAf, sWf, tid, lane, mtA,ntA,mtB,ntB, st, fn);
    }
    // GEMM3: Y = (Y + dist1*W_da^T + b_da) * emask
    {
      auto st = [&](int rr, int col) -> __bf16 { return (__bf16)sD[rr][col]; };
      auto fn = [&](v8f acc, int mt, int nt) {
#pragma unroll
        for (int v = 0; v < 8; ++v) {
          int m = mt*16 + hf*8 + v, n = nt*16 + ln;
          float em = sMaskj[m] * mi;
          sY[m][n] = (sY[m][n] + acc[v] + sVec[5][n]) * em;
        }
      };
      edge_gemm(W_da, sAf, sWf, tid, lane, mtA,ntA,mtB,ntB, st, fn);
    }
    __syncthreads();
    // stats for Y (etoy, emask candidates)
    {
      int c = tid & 63, part = tid >> 6;
      float s=0.f,q=0.f,mn=1e30f,mx=-1e30f;
#pragma unroll 4
      for (int r = 0; r < 16; ++r) {
        int j = part*16 + r;
        float v = sY[j][c];
        float em = sMaskj[j] * mi;
        s += v; q += v*v;
        mn = fminf(mn, v + 1e5f*(1.f-em));
        mx = fmaxf(mx, v - 1e5f*(1.f-em));
      }
      stat_accum(s,q,mn,mx, 8, sRed, sStat, tid);
    }
    // stats for Enew (recomputed inline) and dist1 (mask_j candidates)
    {
      int c = tid & 63, part = tid >> 6;
      float sE=0.f,qE=0.f,mnE=1e30f,mxE=-1e30f;
      float sDd=0.f,qD=0.f,mnD=1e30f,mxD=-1e30f;
#pragma unroll 4
      for (int r = 0; r < 16; ++r) {
        int j = part*16 + r;
        float mj = sMaskj[j]; float em = mj * mi;
        float yv = sY[j][c];
        float en = (yv + sVec[1][c] + yv*sVec[2][c]) * em;
        sE += en; qE += en*en;
        mnE = fminf(mnE, en + 1e5f*(1.f-mj));
        mxE = fmaxf(mxE, en - 1e5f*(1.f-mj));
        float dv = sD[j][c];
        sDd += dv; qD += dv*dv;
        mnD = fminf(mnD, dv + 1e5f*(1.f-mj));
        mxD = fmaxf(mxD, dv - 1e5f*(1.f-mj));
      }
      stat_accum(sE,qE,mnE,mxE, 0, sRed, sStat, tid);
      stat_accum(sDd,qD,mnD,mxD, 4, sRed, sStat, tid);
    }
    // GEMM4: Eout = Enew * W_eo^T + b_eo  (Enew built in A-staging) -> global
    {
      auto st = [&](int rr, int col) -> __bf16 {
        float yv = sY[rr][col]; float em = sMaskj[rr] * mi;
        return (__bf16)((yv + sVec[1][col] + yv*sVec[2][col]) * em);
      };
      auto fn = [&](v8f acc, int mt, int nt) {
#pragma unroll
        for (int v = 0; v < 8; ++v) {
          int m = mt*16 + hf*8 + v, n = nt*16 + ln;
          float em = sMaskj[m] * mi;
          Eout[eBase + (long)m*64 + n] = (acc[v] + sVec[7][n]) * em;
        }
      };
      edge_gemm(W_eo, sAf, sWf, tid, lane, mtA,ntA,mtB,ntB, st, fn);
    }
    __syncthreads();
    // attention logit modulation (RMW on logits buffer)
#pragma unroll
    for (int p2 = 0; p2 < 2; ++p2) {
      int pr = tid + p2*256; int j = pr >> 3, hh = pr & 7;
      float mj = sMaskj[j]; float em = mj * mi;
      float eatt = sAttB[hh], patt = sAttB[8+hh];
      for (int c = 0; c < 64; ++c) {
        float yv = sY[j][c];
        float en = (yv + sVec[1][c] + yv*sVec[2][c]) * em;
        eatt += en * sWatt[hh][c];
        patt += sD[j][c] * sWatt[8+hh][c];
      }
      long lidx = (((long)bi)*256 + jc + j)*8 + hh;
      float a0v = logits[lidx] * em;
      float av = a0v * (1.f + eatt) * (1.f + patt) * em;
      logits[lidx] = (mj > 0.f) ? av : -(__builtin_inff());
    }
    // GEMM5: msg = relu(Y*W_p1^T) . w_p2   (wave shfl row-reduce)
    {
      auto st = [&](int rr, int col) -> __bf16 { return (__bf16)sY[rr][col]; };
      auto fn = [&](v8f acc, int mt, int nt) {
#pragma unroll
        for (int v = 0; v < 8; ++v) {
          int n = nt*16 + ln;
          float x = fmaxf(acc[v], 0.f) * sVec[3][n];
          x += __shfl_xor(x, 1, 32);
          x += __shfl_xor(x, 2, 32);
          x += __shfl_xor(x, 4, 32);
          x += __shfl_xor(x, 8, 32);
          if (ln == 0) sMsgP[mt*16 + hf*8 + v][nt] = x;
        }
      };
      edge_gemm(W_p1, sAf, sWf, tid, lane, mtA,ntA,mtB,ntB, st, fn);
    }
    __syncthreads();
    if (tid < 64) sMsg[tid] = sMsgP[tid][0]+sMsgP[tid][1]+sMsgP[tid][2]+sMsgP[tid][3];
    __syncthreads();
    if (tid < 3) {
      for (int j = 0; j < 64; ++j)
        velAcc += sMsg[j] * (pia - sPosj[j][tid]);
    }
  }
  __syncthreads();
  for (int r = 0; r < 3; ++r) {
    int idx = r*256 + tid;
    if (idx < 768) {
      int c = idx / 12, s = idx % 12;
      rowStats[((long)bi*64 + c)*12 + s] = sStat[c][s];
    }
  }
  if (tid < 3) velRaw[bi*3 + tid] = velAcc * mi;
}

// ---------------------------------------------------------------------------
// Softmax over j per (b,i,h): one wave per head.
// ---------------------------------------------------------------------------
__global__ __launch_bounds__(256)
void k_softmax(float* __restrict__ logits)
{
  int bi = blockIdx.x;
  int wave = threadIdx.x >> 5, lane = threadIdx.x & 31;
  float* p = logits + ((long)bi*256)*8 + wave;
  float v[8]; float mx = -1e30f;
#pragma unroll
  for (int r = 0; r < 8; ++r) { v[r] = p[(lane + r*32)*8]; mx = fmaxf(mx, v[r]); }
  for (int m = 1; m < 32; m <<= 1) mx = fmaxf(mx, __shfl_xor(mx, m, 32));
  float e[8]; float s = 0.f;
#pragma unroll
  for (int r = 0; r < 8; ++r) {
    float ev = (v[r] <= -1e30f) ? 0.f : __expf(v[r] - mx);
    e[r] = ev; s += ev;
  }
  for (int m = 1; m < 32; m <<= 1) s += __shfl_xor(s, m, 32);
  float inv = (s > 0.f) ? 1.f/s : 0.f;
#pragma unroll
  for (int r = 0; r < 8; ++r) p[(lane + r*32)*8] = e[r]*inv;
}

// etox features from per-row stats: [m | min | max | std] per 64-ch block
__global__ void k_feats(const float* __restrict__ rowStats,
                        const float* __restrict__ divA,
                        float* __restrict__ featE, float* __restrict__ featD)
{
  int bi = blockIdx.x; int c = threadIdx.x; int b = bi >> 8;
  const float* st = rowStats + ((long)bi*64 + c)*12;
  float inv = 1.f / divA[b];
  float m = st[0]*inv;
  featE[(long)bi*256 +       c] = m;
  featE[(long)bi*256 +  64 + c] = st[2];
  featE[(long)bi*256 + 128 + c] = st[3];
  featE[(long)bi*256 + 192 + c] = st[1]*inv - m*m;
  float md = st[4]*inv;
  featD[(long)bi*256 +       c] = md;
  featD[(long)bi*256 +  64 + c] = st[6];
  featD[(long)bi*256 + 128 + c] = st[7];
  featD[(long)bi*256 + 192 + c] = st[5]*inv - md*md;
}

// etoy: combine row stats over i, then small 256->64 linears (e_y, dist_y)
__global__ __launch_bounds__(256)
void k_etoy(const float* __restrict__ rowStats, const float* __restrict__ divA,
            const float* __restrict__ W_ey, const float* __restrict__ b_ey,
            const float* __restrict__ W_dy, const float* __restrict__ b_dy,
            float* __restrict__ eyY, float* __restrict__ eyD)
{
  __shared__ float fY[4][256], fD[4][256];
  int tid = threadIdx.x; int b = tid >> 6, c = tid & 63;
  float sy=0.f,qy=0.f,mny=1e30f,mxy=-1e30f;
  float sd=0.f,qd=0.f,mnd=1e30f,mxd=-1e30f;
  for (int i = 0; i < 256; ++i) {
    const float* st = rowStats + ((long)(b*256+i)*64 + c)*12;
    sy += st[8]; qy += st[9]; mny = fminf(mny, st[10]); mxy = fmaxf(mxy, st[11]);
    sd += st[4]; qd += st[5]; mnd = fminf(mnd, st[6]);  mxd = fmaxf(mxd, st[7]);
  }
  float dv = divA[b]; float inv = 1.f/(dv*dv);
  float mY = sy*inv, mD = sd*inv;
  fY[b][c] = mY; fY[b][64+c] = mny; fY[b][128+c] = mxy; fY[b][192+c] = qy*inv - mY*mY;
  fD[b][c] = mD; fD[b][64+c] = mnd; fD[b][128+c] = mxd; fD[b][192+c] = qd*inv - mD*mD;
  __syncthreads();
  float aY = b_ey[c], aD = b_dy[c];
  for (int k = 0; k < 256; ++k) {
    aY += fY[b][k]*W_ey[c*256+k];
    aD += fD[b][k]*W_dy[c*256+k];
  }
  eyY[b*64+c] = aY; eyD[b*64+c] = aD;
}

__global__ void k_newx(const float* __restrict__ outB,
                       const float* __restrict__ mE, const float* __restrict__ mD,
                       const float* __restrict__ yxm, const float* __restrict__ yxa,
                       float* __restrict__ newX)
{
  long idx = (long)blockIdx.x*256 + threadIdx.x;
  int bi = (int)(idx >> 8); int c = (int)(idx & 255); int b = bi >> 8;
  float wv = outB[idx];
  wv *= (1.f + mE[idx]);
  wv *= (1.f + mD[idx]);
  newX[idx] = wv*(yxm[b*256+c] + 1.f) + yxa[b*256+c];
}

__global__ __launch_bounds__(256)
void k_xtoy(const float* __restrict__ newX, const float* __restrict__ mask,
            const float* __restrict__ divA, float* __restrict__ featX)
{
  int b = blockIdx.x; int c = threadIdx.x;
  float s=0.f,q=0.f,mn=1e30f,mx=-1e30f;
  for (int i = 0; i < 256; ++i) {
    float mk = mask[b*256+i];
    float x = newX[((long)(b*256+i))*256 + c]*mk;
    s += x; q += x*x;
    mn = fminf(mn, x + 1e5f*(1.f-mk));
    mx = fmaxf(mx, x - 1e5f*(1.f-mk));
  }
  float dv = divA[b]; float m = s/dv;
  featX[b*1024 +        c] = m;
  featX[b*1024 +  256 + c] = mn;
  featX[b*1024 +  512 + c] = mx;
  featX[b*1024 +  768 + c] = q/dv - m*m;
}

__global__ __launch_bounds__(256)
void k_ypath(const float* __restrict__ yy, const float* __restrict__ eyY,
             const float* __restrict__ eyD, const float* __restrict__ featX,
             const float* __restrict__ W_xy, const float* __restrict__ b_xy,
             const float* __restrict__ W1, const float* __restrict__ b1,
             const float* __restrict__ W2, const float* __restrict__ b2,
             float* __restrict__ yOut)
{
  __shared__ float sNy[4][64], sT[4][64];
  int tid = threadIdx.x; int b = tid >> 6, o = tid & 63;
  float acc = yy[b*64+o] + eyY[b*64+o] + eyD[b*64+o] + b_xy[o];
  for (int k = 0; k < 1024; ++k) acc += featX[b*1024+k]*W_xy[o*1024+k];
  sNy[b][o] = acc; __syncthreads();
  float t = b1[o];
  for (int k = 0; k < 64; ++k) t += sNy[b][k]*W1[o*64+k];
  sT[b][o] = fmaxf(t, 0.f); __syncthreads();
  float yo = b2[o];
  for (int k = 0; k < 64; ++k) yo += sT[b][k]*W2[o*64+k];
  yOut[b*64+o] = yo;
}

__global__ __launch_bounds__(256)
void k_vel(const float* __restrict__ velRaw, const float* __restrict__ mask,
           const float* __restrict__ divA, float* __restrict__ velOut)
{
  __shared__ float red[256];
  int b = blockIdx.x; int i = threadIdx.x;
  float v[3], s[3];
#pragma unroll
  for (int ax = 0; ax < 3; ++ax) v[ax] = velRaw[(b*256+i)*3 + ax];
#pragma unroll
  for (int ax = 0; ax < 3; ++ax) {
    red[i] = v[ax]; __syncthreads();
    for (int off = 128; off > 0; off >>= 1) {
      if (i < off) red[i] += red[i+off];
      __syncthreads();
    }
    s[ax] = red[0]; __syncthreads();
  }
  float dv = divA[b]; float mk = mask[b*256+i];
#pragma unroll
  for (int ax = 0; ax < 3; ++ax)
    velOut[(b*256+i)*3 + ax] = v[ax] - s[ax]/dv*mk;
}

// ---------------------------------------------------------------------------
extern "C" void kernel_launch(void* const* d_in, const int* in_sizes, int n_in,
                              void* d_out, int out_size, void* d_ws, size_t ws_size,
                              hipStream_t stream)
{
  (void)in_sizes; (void)n_in; (void)out_size; (void)ws_size;
  const float* X    = (const float*)d_in[0];
  const float* E    = (const float*)d_in[1];
  const float* y    = (const float*)d_in[2];
  const float* pos  = (const float*)d_in[3];
  const float* mask = (const float*)d_in[4];
  int p = 5;
#define PW(nm) const float* nm = (const float*)d_in[p++]
  PW(W_inE); PW(b_inE);
  PW(W_xe1); PW(b_xe1);
  PW(W_xe2); PW(b_xe2);
  PW(W_dd);  PW(b_dd);
  PW(W_da);  PW(b_da);
  PW(W_dm);  PW(b_dm);
  PW(W_q);   PW(b_q);
  PW(W_k);   PW(b_k);
  PW(W_v);   PW(b_v);
  PW(W_a);
  PW(W_out); PW(b_out);
  PW(W_ea);  PW(b_ea);
  PW(W_pa);  PW(b_pa);
  PW(W_exm); PW(b_exm);
  PW(W_pxm); PW(b_pxm);
  PW(W_yem); PW(b_yem);
  PW(W_yea); PW(b_yea);
  PW(W_yxm); PW(b_yxm);
  PW(W_yxa); PW(b_yxa);
  PW(W_yy);  PW(b_yy);
  PW(W_xy);  PW(b_xy);
  PW(W_ey);  PW(b_ey);
  PW(W_dy);  PW(b_dy);
  PW(W_p1);  PW(W_p2);
  PW(W_xo);  PW(b_xo);
  PW(W_eo);  PW(b_eo);
  PW(W_y1);  PW(b_y1);
  PW(W_y2);  PW(b_y2);
#undef PW

  float* ws = (float*)d_ws;
  float* wsQ     = ws; ws += 262144;
  float* wsK     = ws; ws += 262144;
  float* wsV     = ws; ws += 262144;
  float* wsXe1   = ws; ws += 65536;
  float* wsXe2   = ws; ws += 65536;
  float* wsQh    = ws; ws += 2097152;
  float* wsLog   = ws; ws += 2097152;
  float* wsStats = ws; ws += 786432;
  float* wsVelR  = ws; ws += 4096;
  float* wsFeatE = ws; ws += 262144;
  float* wsFeatD = ws; ws += 262144;
  float* wsModsE = ws; ws += 262144;
  float* wsModsD = ws; ws += 262144;
  float* wsWV    = ws; ws += 2097152;
  float* wsOutB  = ws; ws += 262144;
  float* wsNewX  = ws; ws += 262144;
  float* yeM  = ws; ws += 256;
  float* yeA  = ws; ws += 256;
  float* yxm  = ws; ws += 1024;
  float* yxa  = ws; ws += 1024;
  float* yy   = ws; ws += 256;
  float* divA = ws; ws += 64;
  float* eyY  = ws; ws += 256;
  float* eyD  = ws; ws += 256;
  float* featX= ws; ws += 4096;

  float* Xout   = (float*)d_out;
  float* EoutP  = Xout + 262144;
  float* yOut   = Xout + 17039360;
  float* velOut = Xout + 17039616;

  dim3 B256(256);
  k_prep<<<1, B256, 0, stream>>>(y, mask, W_yem, b_yem, W_yea, b_yea,
                                 W_yxm, b_yxm, W_yxa, b_yxa, W_yy, b_yy,
                                 yeM, yeA, yxm, yxa, yy, divA);
  k_gemm<<<dim3(16,4), B256, 0, stream>>>(X, W_q, b_q, mask, wsQ, 1024,256,256,0);
  k_gemm<<<dim3(16,4), B256, 0, stream>>>(X, W_k, b_k, mask, wsK, 1024,256,256,0);
  k_gemm<<<dim3(16,4), B256, 0, stream>>>(X, W_v, b_v, mask, wsV, 1024,256,256,0);
  k_gemm<<<dim3(16,1), B256, 0, stream>>>(X, W_xe1, b_xe1, mask, wsXe1, 1024,64,256,0);
  k_gemm<<<dim3(16,1), B256, 0, stream>>>(X, W_xe2, b_xe2, mask, wsXe2, 1024,64,256,0);
  k_qh<<<1024, B256, 0, stream>>>(wsQ, W_a, wsQh);
  k_qk<<<dim3(4,8,4), B256, 0, stream>>>(wsQh, wsK, wsLog);
  k_edge<<<1024, B256, 0, stream>>>(E, pos, mask, wsXe1, wsXe2,
                                    W_inE, b_inE, W_dd, b_dd, W_da, b_da,
                                    W_dm, b_dm, W_eo, b_eo, W_ea, b_ea,
                                    W_pa, b_pa, W_p1, W_p2, yeA, yeM,
                                    EoutP, wsLog, wsStats, wsVelR);
  k_softmax<<<1024, B256, 0, stream>>>(wsLog);
  k_wvh<<<dim3(4,8,4), B256, 0, stream>>>(wsLog, wsV, wsWV);
  k_feats<<<1024, 64, 0, stream>>>(wsStats, divA, wsFeatE, wsFeatD);
  k_gemm<<<dim3(16,4), B256, 0, stream>>>(wsFeatE, W_exm, b_exm, (const float*)nullptr,
                                          wsModsE, 1024,256,256,0);
  k_gemm<<<dim3(16,4), B256, 0, stream>>>(wsFeatD, W_pxm, b_pxm, (const float*)nullptr,
                                          wsModsD, 1024,256,256,0);
  k_gemm<<<dim3(16,4), B256, 0, stream>>>(wsWV, W_out, b_out, mask, wsOutB,
                                          1024,256,2048,0);
  k_newx<<<1024, B256, 0, stream>>>(wsOutB, wsModsE, wsModsD, yxm, yxa, wsNewX);
  k_gemm<<<dim3(16,4), B256, 0, stream>>>(wsNewX, W_xo, b_xo, mask, Xout,
                                          1024,256,256,0);
  k_etoy<<<1, B256, 0, stream>>>(wsStats, divA, W_ey, b_ey, W_dy, b_dy, eyY, eyD);
  k_xtoy<<<4, B256, 0, stream>>>(wsNewX, mask, divA, featX);
  k_ypath<<<1, B256, 0, stream>>>(yy, eyY, eyD, featX, W_xy, b_xy,
                                  W_y1, b_y1, W_y2, b_y2, yOut);
  k_vel<<<4, B256, 0, stream>>>(wsVelR, mask, divA, velOut);
}
